// ProposalEncodingLayerV1_37443524886584
// MI455X (gfx1250) — matrix operand order repaired and checked by reading, not compiled, not run on hardware
//
#include <hip/hip_runtime.h>
#include <stdint.h>

typedef __attribute__((ext_vector_type(16))) _Float16 v16h;
typedef __attribute__((ext_vector_type(8)))  _Float16 v8h;
typedef __attribute__((ext_vector_type(4)))  _Float16 v4h;
typedef __attribute__((ext_vector_type(8)))  float    v8f;
typedef __attribute__((ext_vector_type(4)))  float    v4f;

#define NQ 256
#define MK 256
#define DD 256
#define EE 128
#define HH 4
#define HD 32
#define PP 64
#define RR 65536      // M*N keys
#define NSEG 64       // flash-decoding key segments
#define SEGK 1024     // keys per segment
#define QTILES 64     // H * (N/16)

union V16U { v16h v; v8h h[2]; };
union V8U  { v8h v; _Float16 e[8]; };

// CDNA5 async DMA: global -> LDS, 16B per lane, tracked by ASYNCcnt.
__device__ __forceinline__ void async_copy_b128(unsigned lds_off, const void* gaddr) {
  asm volatile("global_load_async_to_lds_b128 %0, %1, off"
               :: "v"(lds_off), "v"((unsigned long long)(uintptr_t)gaddr)
               : "memory");
}
__device__ __forceinline__ void wait_async0() {
  asm volatile("s_wait_asynccnt 0x0" ::: "memory");
}

// ---------------------------------------------------------------------------
// xPhi = x @ phiD.T, yPhi = y @ phiD.T, xG = x @ gD.T, yG = y @ gD.T
// grid 1024 x 128
__global__ void pe_proj_kernel(const float* __restrict__ x, const float* __restrict__ y,
                               const float* __restrict__ phi_w, const float* __restrict__ g_w,
                               float* __restrict__ xPhi, float* __restrict__ yPhi,
                               float* __restrict__ xG, float* __restrict__ yG) {
  const int which = blockIdx.x >> 8;
  const int n = blockIdx.x & 255;
  const int e = threadIdx.x;
  const float* src = (which & 1) ? (y + n * DD) : (x + n * DD);
  const float* wt  = ((which < 2) ? phi_w : g_w) + e * (DD + EE);
  const v4f* s4 = (const v4f*)src;
  const v4f* w4 = (const v4f*)wt;
  float acc = 0.f;
  for (int d = 0; d < DD / 4; ++d) {
    const v4f a = s4[d], b = w4[d];
    acc += a[0] * b[0] + a[1] * b[1] + a[2] * b[2] + a[3] * b[3];
  }
  float* dst = (which == 0) ? xPhi : (which == 1) ? yPhi : (which == 2) ? xG : yG;
  dst[n * EE + e] = acc;
}

// ---------------------------------------------------------------------------
// Wk[e][p] = sum_j phi_w[e,256+j]*k_w2[j,p] (f16); same for Wg with g_w.
// kbias[e] = phiE[e,:] . k_b2 ; gbias likewise.  grid 128 x 128
__global__ void pe_wfold_kernel(const float* __restrict__ phi_w, const float* __restrict__ g_w,
                                const float* __restrict__ k_w2, const float* __restrict__ k_b2,
                                _Float16* __restrict__ Wk, _Float16* __restrict__ Wg,
                                float* __restrict__ kbias, float* __restrict__ gbias) {
  const int e = blockIdx.x;
  const int t = threadIdx.x;
  const int p = t & 63;
  const float* base = (t < 64) ? phi_w : g_w;
  float acc = 0.f;
  for (int j = 0; j < EE; ++j) acc += base[e * (DD + EE) + DD + j] * k_w2[j * PP + p];
  if (t < 64) Wk[e * PP + p] = (_Float16)acc; else Wg[e * PP + p] = (_Float16)acc;
  if (t == 0) {
    float bk = 0.f, bg = 0.f;
    for (int j = 0; j < EE; ++j) {
      bk += phi_w[e * (DD + EE) + DD + j] * k_b2[j];
      bg += g_w[e * (DD + EE) + DD + j] * k_b2[j];
    }
    kbias[e] = bk; gbias[e] = bg;
  }
}

// ---------------------------------------------------------------------------
// q = concat(x, MLP(x_pos)) @ theta_w.T, scaled, to qh[h][n][hd] f16.
// grid 256 x 128
__global__ void pe_q_kernel(const float* __restrict__ x, const float* __restrict__ x_pos,
                            const float* __restrict__ q_w1, const float* __restrict__ q_b1,
                            const float* __restrict__ q_w2, const float* __restrict__ q_b2,
                            const float* __restrict__ theta_w, _Float16* __restrict__ qh) {
  __shared__ float hq[PP];
  __shared__ float qpe[EE];
  const int n = blockIdx.x;
  const int t = threadIdx.x;
  if (t < PP) {
    float a = q_b1[t] + x_pos[n * 3] * q_w1[t * 3] + x_pos[n * 3 + 1] * q_w1[t * 3 + 1] +
              x_pos[n * 3 + 2] * q_w1[t * 3 + 2];
    hq[t] = fmaxf(a, 0.f);
  }
  __syncthreads();
  float acc = q_b2[t];
  {
    const v4f* h4 = (const v4f*)hq;
    const v4f* w4 = (const v4f*)(q_w2 + t * PP);
    for (int p = 0; p < PP / 4; ++p) {
      const v4f a = h4[p], b = w4[p];
      acc += a[0] * b[0] + a[1] * b[1] + a[2] * b[2] + a[3] * b[3];
    }
  }
  qpe[t] = acc;
  __syncthreads();
  float qv = 0.f;
  const float* wt = theta_w + t * (DD + EE);
  {
    const v4f* x4 = (const v4f*)(x + n * DD);
    const v4f* w4 = (const v4f*)wt;
    for (int d = 0; d < DD / 4; ++d) {
      const v4f a = x4[d], b = w4[d];
      qv += a[0] * b[0] + a[1] * b[1] + a[2] * b[2] + a[3] * b[3];
    }
    const v4f* q4 = (const v4f*)qpe;
    const v4f* w4b = (const v4f*)(wt + DD);
    for (int j = 0; j < EE / 4; ++j) {
      const v4f a = q4[j], b = w4b[j];
      qv += a[0] * b[0] + a[1] * b[1] + a[2] * b[2] + a[3] * b[3];
    }
  }
  qv *= 0.17677669529663687f;  // HD^-0.5
  qh[(size_t)((t >> 5) * NQ + n) * HD + (t & 31)] = (_Float16)qv;
}

// ---------------------------------------------------------------------------
// K/V build with WMMA.  grid 1024 x 128 (4 waves, 16 rows each).
// K_out[h][r][hd] f16, Vt_out[h][hd][r] f16.
__global__ void kv_build_kernel(const float* __restrict__ x_pos, const float* __restrict__ y_pos,
                                const float* __restrict__ k_w1, const float* __restrict__ k_b1,
                                const float* __restrict__ xPhi, const float* __restrict__ yPhi,
                                const float* __restrict__ xG, const float* __restrict__ yG,
                                const _Float16* __restrict__ Wk, const _Float16* __restrict__ Wg,
                                const float* __restrict__ kbias, const float* __restrict__ gbias,
                                _Float16* __restrict__ K_out, _Float16* __restrict__ Vt_out) {
  __shared__ _Float16 wk_s[EE * PP];
  __shared__ _Float16 wg_s[EE * PP];
  __shared__ _Float16 h_s[4][16 * PP];
  __shared__ _Float16 kt_s[4][16 * 16];
  const int tid = threadIdx.x;
  // Stage folded weights into LDS via CDNA5 async DMA (16B/lane chunks).
  {
    const unsigned lwk = (unsigned)(uintptr_t)wk_s;
    const unsigned lwg = (unsigned)(uintptr_t)wg_s;
    const char* gwk = (const char*)Wk;
    const char* gwg = (const char*)Wg;
    for (int i = tid; i < EE * PP / 8; i += 128) {
      async_copy_b128(lwk + i * 16, gwk + i * 16);
      async_copy_b128(lwg + i * 16, gwg + i * 16);
    }
    wait_async0();
  }
  const int w = tid >> 5, lane = tid & 31;
  const int r0 = (blockIdx.x * 4 + w) * 16;
  const int m = r0 >> 8, n0 = r0 & 255;   // 16 rows share m
  const float yp0 = y_pos[m * 3], yp1 = y_pos[m * 3 + 1], yp2 = y_pos[m * 3 + 2];
  for (int it = 0; it < 32; ++it) {
    const int idx = lane + 32 * it;       // 16*64 values
    const int row = idx >> 6, p = idx & 63;
    const int nn = n0 + row;
    const float d0 = x_pos[nn * 3] - yp0;
    const float d1 = x_pos[nn * 3 + 1] - yp1;
    const float d2 = x_pos[nn * 3 + 2] - yp2;
    float hv = k_b1[p] + d0 * k_w1[p * 3] + d1 * k_w1[p * 3 + 1] + d2 * k_w1[p * 3 + 2];
    h_s[w][row * PP + p] = (_Float16)fmaxf(hv, 0.f);
  }
  __syncthreads();
  const int l15 = lane & 15;
  const int hi8 = (lane < 16) ? 0 : 8;
  const int hi16 = (lane < 16) ? 0 : 16;
  const int rowBase = (lane < 16) ? 0 : 8;
  V16U a0, a1;
  {
    const _Float16* hr = &h_s[w][l15 * PP];
    a0.h[0] = *(const v8h*)(hr + hi8);       a0.h[1] = *(const v8h*)(hr + 16 + hi8);
    a1.h[0] = *(const v8h*)(hr + 32 + hi8);  a1.h[1] = *(const v8h*)(hr + 48 + hi8);
  }
  for (int t = 0; t < 8; ++t) {
    const int eL = t * 16 + l15;
    const int hHead = t >> 1;
    const int hd0 = (t & 1) * 16;         // tile = contiguous halves hd0..hd0+15 of K rows
    // Batch all epilogue loads first (one clause, no serialization).
    float xphi_r[8], xg_r[8];
    for (int r = 0; r < 8; ++r) {
      const int nn = n0 + rowBase + r;
      xphi_r[r] = xPhi[nn * EE + eL];
      xg_r[r]   = xG[nn * EE + eL];
    }
    const float kadd = kbias[eL] - yPhi[m * EE + eL];
    const float gadd = gbias[eL] - yG[m * EE + eL];
    V16U b0, b1;
    {
      const _Float16* wr = &wk_s[eL * PP];
      b0.h[0] = *(const v8h*)(wr + hi8);       b0.h[1] = *(const v8h*)(wr + 16 + hi8);
      b1.h[0] = *(const v8h*)(wr + 32 + hi8);  b1.h[1] = *(const v8h*)(wr + 48 + hi8);
    }
    v8f dk = {};
    dk = __builtin_amdgcn_wmma_f32_16x16x32_f16(false, a0.v, false, b0.v, (short)0, dk, false, false);
    dk = __builtin_amdgcn_wmma_f32_16x16x32_f16(false, a1.v, false, b1.v, (short)0, dk, false, false);
    {
      const _Float16* wr = &wg_s[eL * PP];
      b0.h[0] = *(const v8h*)(wr + hi8);       b0.h[1] = *(const v8h*)(wr + 16 + hi8);
      b1.h[0] = *(const v8h*)(wr + 32 + hi8);  b1.h[1] = *(const v8h*)(wr + 48 + hi8);
    }
    v8f dv = {};
    dv = __builtin_amdgcn_wmma_f32_16x16x32_f16(false, a0.v, false, b0.v, (short)0, dv, false, false);
    dv = __builtin_amdgcn_wmma_f32_16x16x32_f16(false, a1.v, false, b1.v, (short)0, dv, false, false);
    // K epilogue: transpose-bounce through LDS -> coalesced b128 stores.
    for (int r = 0; r < 8; ++r)
      kt_s[w][(rowBase + r) * 16 + l15] = (_Float16)(dk[r] + xphi_r[r] + kadd);
    __builtin_amdgcn_wave_barrier();
    {
      const int row = l15, half = lane >> 4;
      V8U kp;
      kp.v = *(const v8h*)&kt_s[w][row * 16 + half * 8];
      *(v8h*)(K_out + ((size_t)hHead * RR + r0 + row) * HD + hd0 + half * 8) = kp.v;
    }
    __builtin_amdgcn_wave_barrier();
    // V epilogue: already contiguous in transposed layout.
    V8U pack;
    for (int r = 0; r < 8; ++r) pack.e[r] = (_Float16)(dv[r] + xg_r[r] + gadd);
    *(v8h*)(Vt_out + ((size_t)(hHead * HD + hd0 + l15)) * RR + r0 + rowBase) = pack.v;
  }
}

// ---------------------------------------------------------------------------
// Flash attention (split-K over segments).  grid 4096 x 32 (1 wave/block).
__global__ void attn_kernel(const _Float16* __restrict__ qh, const _Float16* __restrict__ Kf,
                            const _Float16* __restrict__ Vt, float* __restrict__ Opart,
                            float* __restrict__ mpart, float* __restrict__ lpart) {
  __shared__ float scores[16][128];
  __shared__ _Float16 pbuf[16][128];
  __shared__ float alpha_s[16];
  const int lane = threadIdx.x;
  const int qt = blockIdx.x & (QTILES - 1);
  const int seg = blockIdx.x >> 6;
  const int h = qt >> 4;
  const int q0 = (qt & 15) * 16;
  const int l15 = lane & 15;
  const int hi8 = (lane < 16) ? 0 : 8;
  const int hi16 = (lane < 16) ? 0 : 16;
  const int rowBase = (lane < 16) ? 0 : 8;
  V16U aq;
  {
    const _Float16* qp = qh + ((size_t)h * NQ + q0 + l15) * HD;
    aq.h[0] = *(const v8h*)(qp + hi8);
    aq.h[1] = *(const v8h*)(qp + 16 + hi8);
  }
  v8f o0 = {}, o1 = {};
  float m_i = -1e30f, l_i = 0.f;
  const int myrow = l15;
  const int chalf = lane >> 4;
  for (int b = 0; b < 8; ++b) {
    const int kb0 = seg * SEGK + b * 128;
    if (b < 7) {  // prefetch next key block (L2-resident streams)
      __builtin_prefetch(Kf + ((size_t)h * RR + kb0 + 128 + (lane << 2)) * HD, 0, 1);
      __builtin_prefetch(Vt + ((size_t)h * HD + lane) * RR + kb0 + 128, 0, 1);
    }
    for (int jt = 0; jt < 8; ++jt) {
      const int j = kb0 + jt * 16 + l15;
      V16U bk;
      const _Float16* kp = Kf + ((size_t)h * RR + j) * HD + hi8;
      bk.h[0] = *(const v8h*)kp;
      bk.h[1] = *(const v8h*)(kp + 16);
      v8f s = {};
      s = __builtin_amdgcn_wmma_f32_16x16x32_f16(false, aq.v, false, bk.v, (short)0, s, false, false);
      for (int r = 0; r < 8; ++r) scores[rowBase + r][jt * 16 + l15] = s[r];
    }
    __syncthreads();
    // Row-wise online softmax (vectorized LDS scans).
    const v4f* srow4 = (const v4f*)&scores[myrow][chalf * 64];
    float bmax = -1e30f;
    for (int c = 0; c < 16; ++c) {
      const v4f s4 = srow4[c];
      bmax = fmaxf(bmax, fmaxf(fmaxf(s4[0], s4[1]), fmaxf(s4[2], s4[3])));
    }
    bmax = fmaxf(bmax, __shfl_xor(bmax, 16, 32));
    const float m_new = fmaxf(m_i, bmax);
    const float alpha = __expf(m_i - m_new);
    float psum = 0.f;
    _Float16* prow = &pbuf[myrow][chalf * 64];
    for (int c = 0; c < 16; ++c) {
      const v4f s4 = srow4[c];
      const float p0 = __expf(s4[0] - m_new), p1 = __expf(s4[1] - m_new);
      const float p2 = __expf(s4[2] - m_new), p3 = __expf(s4[3] - m_new);
      psum += (p0 + p1) + (p2 + p3);
      v4h ph;
      ph[0] = (_Float16)p0; ph[1] = (_Float16)p1; ph[2] = (_Float16)p2; ph[3] = (_Float16)p3;
      *(v4h*)(prow + c * 4) = ph;
    }
    psum += __shfl_xor(psum, 16, 32);
    l_i = alpha * l_i + psum;
    m_i = m_new;
    if (lane < 16) alpha_s[myrow] = alpha;
    __syncthreads();
    for (int r = 0; r < 8; ++r) {
      const float a = alpha_s[rowBase + r];
      o0[r] *= a;  o1[r] *= a;
    }
    for (int kc = 0; kc < 4; ++kc) {
      V16U ap;
      const _Float16* pr = &pbuf[l15][kc * 32];
      ap.h[0] = *(const v8h*)(pr + hi8);
      ap.h[1] = *(const v8h*)(pr + 16 + hi8);
      const int krow = kb0 + kc * 32 + hi8;
      {
        const _Float16* vp = Vt + ((size_t)(h * HD + l15)) * RR + krow;
        V16U bv; bv.h[0] = *(const v8h*)vp; bv.h[1] = *(const v8h*)(vp + 16);
        o0 = __builtin_amdgcn_wmma_f32_16x16x32_f16(false, ap.v, false, bv.v, (short)0, o0, false, false);
      }
      {
        const _Float16* vp = Vt + ((size_t)(h * HD + 16 + l15)) * RR + krow;
        V16U bv; bv.h[0] = *(const v8h*)vp; bv.h[1] = *(const v8h*)(vp + 16);
        o1 = __builtin_amdgcn_wmma_f32_16x16x32_f16(false, ap.v, false, bv.v, (short)0, o1, false, false);
      }
    }
    __syncthreads();
  }
  float* op = Opart + ((size_t)(qt * NSEG + seg) * 16) * HD;
  for (int r = 0; r < 8; ++r) {
    op[(rowBase + r) * HD + l15] = o0[r];
    op[(rowBase + r) * HD + 16 + l15] = o1[r];
  }
  if (lane < 16) {
    mpart[(qt * NSEG + seg) * 16 + myrow] = m_i;
    lpart[(qt * NSEG + seg) * 16 + myrow] = l_i;
  }
}

// ---------------------------------------------------------------------------
// Merge segment partials.  grid 64 x 256
__global__ void combine_kernel(const float* __restrict__ Opart, const float* __restrict__ mpart,
                               const float* __restrict__ lpart, float* __restrict__ Ofull) {
  const int qt = blockIdx.x;
  const int h = qt >> 4, nt = qt & 15;
  for (int idx = threadIdx.x; idx < 16 * HD; idx += blockDim.x) {
    const int q = idx >> 5, hd = idx & 31;
    float M = -1e30f;
    for (int s = 0; s < NSEG; ++s) M = fmaxf(M, mpart[(qt * NSEG + s) * 16 + q]);
    float L = 0.f, acc = 0.f;
    for (int s = 0; s < NSEG; ++s) {
      const float wgt = __expf(mpart[(qt * NSEG + s) * 16 + q] - M);
      L += wgt * lpart[(qt * NSEG + s) * 16 + q];
      acc += wgt * Opart[((size_t)(qt * NSEG + s) * 16 + q) * HD + hd];
    }
    Ofull[(nt * 16 + q) * EE + h * HD + hd] = acc / L;
  }
}

// ---------------------------------------------------------------------------
// out projection + residual + LayerNorm.  grid 256 x 256
__global__ void out_ln_kernel(const float* __restrict__ x, const float* __restrict__ Ofull,
                              const float* __restrict__ out_w, const float* __restrict__ ln_g,
                              const float* __restrict__ ln_b, float* __restrict__ out) {
  __shared__ float orow[EE];
  __shared__ float red[DD];
  const int n = blockIdx.x, d = threadIdx.x;
  if (d < EE) orow[d] = Ofull[n * EE + d];
  __syncthreads();
  float acc = x[n * DD + d];
  {
    const v4f* o4 = (const v4f*)orow;
    const v4f* w4 = (const v4f*)(out_w + d * EE);
    for (int e = 0; e < EE / 4; ++e) {
      const v4f a = o4[e], b = w4[e];
      acc += a[0] * b[0] + a[1] * b[1] + a[2] * b[2] + a[3] * b[3];
    }
  }
  red[d] = acc;
  __syncthreads();
  for (int s = DD / 2; s > 0; s >>= 1) { if (d < s) red[d] += red[d + s]; __syncthreads(); }
  const float mu = red[0] / (float)DD;
  __syncthreads();
  const float c = acc - mu;
  red[d] = c * c;
  __syncthreads();
  for (int s = DD / 2; s > 0; s >>= 1) { if (d < s) red[d] += red[d + s]; __syncthreads(); }
  const float var = red[0] / (float)DD;
  out[n * DD + d] = c * rsqrtf(var + 1e-5f) * ln_g[d] + ln_b[d];
}

// ---------------------------------------------------------------------------
extern "C" void kernel_launch(void* const* d_in, const int* in_sizes, int n_in,
                              void* d_out, int out_size, void* d_ws, size_t ws_size,
                              hipStream_t stream) {
  (void)in_sizes; (void)n_in; (void)out_size; (void)ws_size;
  const float* x       = (const float*)d_in[0];
  const float* y       = (const float*)d_in[1];
  const float* x_pos   = (const float*)d_in[2];
  const float* y_pos   = (const float*)d_in[3];
  const float* theta_w = (const float*)d_in[4];
  const float* phi_w   = (const float*)d_in[5];
  const float* g_w     = (const float*)d_in[6];
  const float* q_w1    = (const float*)d_in[7];
  const float* q_b1    = (const float*)d_in[8];
  const float* q_w2    = (const float*)d_in[9];
  const float* q_b2    = (const float*)d_in[10];
  const float* k_w1    = (const float*)d_in[11];
  const float* k_b1    = (const float*)d_in[12];
  const float* k_w2    = (const float*)d_in[13];
  const float* k_b2    = (const float*)d_in[14];
  const float* out_w   = (const float*)d_in[15];
  const float* ln_g    = (const float*)d_in[16];
  const float* ln_b    = (const float*)d_in[17];
  float* out = (float*)d_out;

  char* ws = (char*)d_ws;
  const size_t OFF_QH   = 0;                          // 4*256*32 f16   = 64 KB
  const size_t OFF_K    = OFF_QH + 65536;             // 4*65536*32 f16 = 16 MB
  const size_t OFF_VT   = OFF_K + 16777216;           // 16 MB
  const size_t OFF_XPHI = OFF_VT + 16777216;          // 256*128 f32
  const size_t OFF_YPHI = OFF_XPHI + 131072;
  const size_t OFF_XG   = OFF_YPHI + 131072;
  const size_t OFF_YG   = OFF_XG + 131072;
  const size_t OFF_WK   = OFF_YG + 131072;            // 128*64 f16
  const size_t OFF_WG   = OFF_WK + 16384;
  const size_t OFF_KB   = OFF_WG + 16384;             // 128 f32
  const size_t OFF_GB   = OFF_KB + 512;
  const size_t OFF_MP   = OFF_GB + 512;               // 64*64*16 f32
  const size_t OFF_LP   = OFF_MP + 262144;
  const size_t OFF_OP   = OFF_LP + 262144;            // 64*64*16*32 f32 = 8 MB
  const size_t OFF_OF   = OFF_OP + 8388608;           // 256*128 f32

  _Float16* qh   = (_Float16*)(ws + OFF_QH);
  _Float16* Kf   = (_Float16*)(ws + OFF_K);
  _Float16* Vt   = (_Float16*)(ws + OFF_VT);
  float* xPhi    = (float*)(ws + OFF_XPHI);
  float* yPhi    = (float*)(ws + OFF_YPHI);
  float* xG      = (float*)(ws + OFF_XG);
  float* yG      = (float*)(ws + OFF_YG);
  _Float16* Wk   = (_Float16*)(ws + OFF_WK);
  _Float16* Wg   = (_Float16*)(ws + OFF_WG);
  float* kbias   = (float*)(ws + OFF_KB);
  float* gbias   = (float*)(ws + OFF_GB);
  float* mpart   = (float*)(ws + OFF_MP);
  float* lpart   = (float*)(ws + OFF_LP);
  float* Opart   = (float*)(ws + OFF_OP);
  float* Ofull   = (float*)(ws + OFF_OF);

  pe_proj_kernel<<<1024, 128, 0, stream>>>(x, y, phi_w, g_w, xPhi, yPhi, xG, yG);
  pe_wfold_kernel<<<128, 128, 0, stream>>>(phi_w, g_w, k_w2, k_b2, Wk, Wg, kbias, gbias);
  pe_q_kernel<<<256, 128, 0, stream>>>(x, x_pos, q_w1, q_b1, q_w2, q_b2, theta_w, qh);
  kv_build_kernel<<<1024, 128, 0, stream>>>(x_pos, y_pos, k_w1, k_b1, xPhi, yPhi, xG, yG,
                                            Wk, Wg, kbias, gbias, Kf, Vt);
  attn_kernel<<<QTILES * NSEG, 32, 0, stream>>>(qh, Kf, Vt, Opart, mpart, lpart);
  combine_kernel<<<QTILES, 256, 0, stream>>>(Opart, mpart, lpart, Ofull);
  out_ln_kernel<<<NQ, DD, 0, stream>>>(x, Ofull, out_w, ln_g, ln_b, out);
}
